// DeepseekMoE_63969242906687
// MI455X (gfx1250) — matrix operand order repaired and checked by reading, not compiled, not run on hardware
//
#include <hip/hip_runtime.h>
#include <hip/hip_bf16.h>
#include <math.h>

// ---------------- problem constants ----------------
#define NE    64      // experts
#define TK    6       // top-k
#define HD    128     // hidden dim
#define MI    80      // moe intermediate
#define MIP   96      // MI padded to multiple of 32 (WMMA K)
#define MS    160     // shared intermediate (5 * 32)
#define NTOK  16384   // B*S tokens
#define TILE  64      // token rows per GEMM tile
#define MAXTILES (NE + (NTOK*TK)/TILE)   // 64 + 1536 = 1600

typedef __attribute__((ext_vector_type(8)))  __bf16 v8bf;
typedef __attribute__((ext_vector_type(16))) __bf16 v16bf;
typedef __attribute__((ext_vector_type(8)))  float  v8f;

// ---- WMMA fragment loads from LDS (wave32 layouts per CDNA5 ISA 7.12.2) ----
// A (16x32 bf16): lane row = lane&15; lanes 0-15 hold K = [k0..k0+7] and [k0+16..k0+23],
// lanes 16-31 hold K = [k0+8..] and [k0+24..].
__device__ __forceinline__ v16bf ld_frag_a(const __bf16* p, int row0, int k0,
                                           int stride, int lane) {
  int r   = row0 + (lane & 15);
  int klo = k0 + ((lane >> 4) << 3);
  v8bf lo = *(const v8bf*)(p + r * stride + klo);
  v8bf hi = *(const v8bf*)(p + r * stride + klo + 16);
  return __builtin_shufflevector(lo, hi, 0,1,2,3,4,5,6,7,8,9,10,11,12,13,14,15);
}
// B (32x16 bf16), stored transposed in LDS as [N][K]: lane col = lane&15;
// lanes 0-15 hold K = [k0..k0+7] and [k0+16..k0+23], lanes 16-31 hold K = [k0+8..] and [k0+24..].
__device__ __forceinline__ v16bf ld_frag_b(const __bf16* p, int col0, int k0,
                                           int stride, int lane) {
  int c  = col0 + (lane & 15);
  int kb = k0 + ((lane >> 4) << 3);
  v8bf lo = *(const v8bf*)(p + c * stride + kb);
  v8bf hi = *(const v8bf*)(p + c * stride + kb + 16);
  return __builtin_shufflevector(lo, hi, 0,1,2,3,4,5,6,7,8,9,10,11,12,13,14,15);
}

// fast silu: g * rcp(1+exp(-g)); v_exp_f32 + v_rcp_f32, no IEEE-div Newton tail.
// exp(-g)->inf => rcp->0 => silu->0 (correct limit).
__device__ __forceinline__ float silu_f(float g) {
  return g * __builtin_amdgcn_rcpf(1.0f + __expf(-g));
}

// ---------------- K0: zero counts ----------------
__global__ void k_init(int* __restrict__ counts) {
  int i = blockIdx.x * blockDim.x + threadIdx.x;
  if (i < NE) counts[i] = 0;
}

// ---------------- K1: gating (one wave32 per token) ----------------
__global__ __launch_bounds__(128) void k_gate(const float* __restrict__ x,
                                              const float* __restrict__ gw,
                                              int* __restrict__ topk_idx,
                                              float* __restrict__ topk_w,
                                              int* __restrict__ counts) {
  __shared__ float xs[4][HD];
  const int lane = threadIdx.x & 31;
  const int wv   = threadIdx.x >> 5;
  const int t    = blockIdx.x * 4 + wv;

  // stage token row into LDS (4 floats / lane)
  #pragma unroll
  for (int j = 0; j < 4; j++)
    xs[wv][lane * 4 + j] = x[(size_t)t * HD + lane * 4 + j];
  __syncthreads();

  // each lane: logits for experts `lane` and `lane+32`
  float a0 = 0.f, a1 = 0.f;
  const float* g0 = gw + (size_t)lane * HD;
  const float* g1 = gw + (size_t)(lane + 32) * HD;
  for (int d = 0; d < HD; d++) {
    float xv = xs[wv][d];
    a0 = fmaf(xv, g0[d], a0);
    a1 = fmaf(xv, g1[d], a1);
  }
  // softmax over 64 logits
  float m = fmaxf(a0, a1);
  for (int o = 16; o; o >>= 1) m = fmaxf(m, __shfl_xor(m, o, 32));
  float e0 = __expf(a0 - m), e1 = __expf(a1 - m);
  float s = e0 + e1;
  for (int o = 16; o; o >>= 1) s += __shfl_xor(s, o, 32);
  float r0 = e0 / s, r1 = e1 / s;

  // iterative top-6 via wave argmax
  float wsum = 0.f, myv = 0.f;
  int myi = 0;
  #pragma unroll
  for (int it = 0; it < TK; it++) {
    float v; int idx;
    if (r1 > r0) { v = r1; idx = lane + 32; } else { v = r0; idx = lane; }
    for (int o = 16; o; o >>= 1) {
      float ov = __shfl_xor(v, o, 32);
      int   oi = __shfl_xor(idx, o, 32);
      if (ov > v || (ov == v && oi < idx)) { v = ov; idx = oi; }
    }
    wsum += v;
    if (lane == it) { myv = v; myi = idx; }
    if (idx == lane) r0 = -1.f;
    else if (idx == lane + 32) r1 = -1.f;
  }
  if (lane < TK) {
    topk_idx[t * TK + lane] = myi;
    topk_w[t * TK + lane]   = myv / (wsum + 1e-20f);
    atomicAdd(&counts[myi], 1);
  }
}

// ---------------- K2: scans ----------------
__global__ void k_scan(const int* __restrict__ counts, int* __restrict__ offsets,
                       int* __restrict__ tileStart, int* __restrict__ cursor) {
  if (threadIdx.x == 0 && blockIdx.x == 0) {
    int off = 0, ts = 0;
    for (int e = 0; e < NE; e++) {
      offsets[e] = off;
      cursor[e]  = off;
      tileStart[e] = ts;
      int c = counts[e];
      off += c;
      ts  += (c + TILE - 1) / TILE;
    }
    offsets[NE]   = off;
    tileStart[NE] = ts;
  }
}

// ---------------- K3: scatter tokens into expert buckets ----------------
__global__ __launch_bounds__(256) void k_scatter(const int* __restrict__ topk_idx,
                                                 const float* __restrict__ topk_w,
                                                 int* __restrict__ cursor,
                                                 int* __restrict__ bucket_t,
                                                 float* __restrict__ bucket_w) {
  int i = blockIdx.x * 256 + threadIdx.x;
  if (i >= NTOK * TK) return;
  int e = topk_idx[i];
  int pos = atomicAdd(&cursor[e], 1);
  bucket_t[pos] = i / TK;
  bucket_w[pos] = topk_w[i];
}

// ---------------- K5: shared experts (writes d_out, plain stores) ----------------
// LDS: A[64][128] Bg[160][128] Bu[160][128] Bd[128][160] H[64][160]  (bf16)
__global__ __launch_bounds__(256) void k_shared(const float* __restrict__ x,
                                                const float* __restrict__ swg,
                                                const float* __restrict__ swu,
                                                const float* __restrict__ swd,
                                                float* __restrict__ out) {
  extern __shared__ __align__(16) char smem[];
  __bf16* A  = (__bf16*)smem;            // 64 x 128
  __bf16* Bg = A  + TILE * HD;           // 160 x 128 (transposed: [m][d])
  __bf16* Bu = Bg + MS * HD;
  __bf16* Bd = Bu + MS * HD;             // 128 x 160 (transposed: [d][m])
  __bf16* H  = Bd + HD * MS;             // 64 x 160

  const int t0   = blockIdx.x * TILE;
  const int tid  = threadIdx.x;
  const int lane = tid & 31;
  const int wv   = tid >> 5;

  for (int i = tid; i < TILE * HD; i += 256) {
    int r = i >> 7, d = i & 127;
    A[r * HD + d] = (__bf16)x[(size_t)(t0 + r) * HD + d];
  }
  for (int i = tid; i < HD * MS; i += 256) {   // swg/swu: [d][m]
    int d = i / MS, mm = i - d * MS;
    Bg[mm * HD + d] = (__bf16)swg[i];
    Bu[mm * HD + d] = (__bf16)swu[i];
  }
  for (int i = tid; i < MS * HD; i += 256) {   // swd: [m][d]
    int mm = i >> 7, d = i & 127;
    Bd[d * MS + mm] = (__bf16)swd[i];
  }
  __syncthreads();

  // GEMM1: H = silu(A@Wg) * (A@Wu) ; 4 m-tiles x 10 n-tiles, K=128 (4 steps)
  for (int task = wv; task < 40; task += 8) {
    int mt = task / 10, nt = task - mt * 10;
    v8f cg = {}; v8f cu = {};
    #pragma unroll
    for (int ks = 0; ks < 4; ks++) {
      v16bf a  = ld_frag_a(A,  mt * 16, ks * 32, HD, lane);
      v16bf bg = ld_frag_b(Bg, nt * 16, ks * 32, HD, lane);
      v16bf bu = ld_frag_b(Bu, nt * 16, ks * 32, HD, lane);
      cg = __builtin_amdgcn_wmma_f32_16x16x32_bf16(false, a, false, bg, (short)0, cg, false, false);
      cu = __builtin_amdgcn_wmma_f32_16x16x32_bf16(false, a, false, bu, (short)0, cu, false, false);
    }
    int rbase = mt * 16 + ((lane >> 4) << 3);
    int col   = nt * 16 + (lane & 15);
    #pragma unroll
    for (int r = 0; r < 8; r++)
      H[(rbase + r) * MS + col] = (__bf16)(silu_f(cg[r]) * cu[r]);
  }
  __syncthreads();

  // GEMM2: OUT = H @ Wd ; 4 m-tiles x 8 n-tiles, K=160 (5 steps)
  for (int task = wv; task < 32; task += 8) {
    int mt = task >> 3, nt = task & 7;
    v8f c = {};
    #pragma unroll
    for (int ks = 0; ks < 5; ks++) {
      v16bf a = ld_frag_a(H,  mt * 16, ks * 32, MS, lane);
      v16bf b = ld_frag_b(Bd, nt * 16, ks * 32, MS, lane);
      c = __builtin_amdgcn_wmma_f32_16x16x32_bf16(false, a, false, b, (short)0, c, false, false);
    }
    int rbase = mt * 16 + ((lane >> 4) << 3);
    int col   = nt * 16 + (lane & 15);
    #pragma unroll
    for (int r = 0; r < 8; r++)
      out[(size_t)(t0 + rbase + r) * HD + col] = c[r];
  }
}

// ---------------- K6: routed experts (grouped GEMM, atomic combine) --------
// LDS: A[64][136] Bg[80][136] Bu[80][136] Bd[128][96] H[64][96] + w[64] + tok[64]
#define AST 136   // A / Bg / Bu K-stride (128 padded, 16B aligned, fewer bank conflicts)
__global__ __launch_bounds__(256) void k_routed(const float* __restrict__ x,
                                                const float* __restrict__ wg,
                                                const float* __restrict__ wu,
                                                const float* __restrict__ wd,
                                                const int* __restrict__ tileStart,
                                                const int* __restrict__ offsets,
                                                const int* __restrict__ bucket_t,
                                                const float* __restrict__ bucket_w,
                                                float* __restrict__ out) {
  extern __shared__ __align__(16) char smem[];
  __bf16* A  = (__bf16*)smem;            // 64 x 136
  __bf16* Bg = A  + TILE * AST;          // 80 x 136  ([m][d])
  __bf16* Bu = Bg + MI * AST;
  __bf16* Bd = Bu + MI * AST;            // 128 x 96  ([d][m], m padded)
  __bf16* H  = Bd + HD * MIP;            // 64 x 96
  float*  wl = (float*)(H + TILE * MIP);
  int*    tl = (int*)(wl + TILE);

  const int b = blockIdx.x;
  if (b >= tileStart[NE]) return;        // inactive tile (uniform exit)
  int e = 0;
  while (tileStart[e + 1] <= b) e++;
  const int row0 = offsets[e] + ((b - tileStart[e]) * TILE);
  int n = offsets[e + 1] - row0;
  if (n > TILE) n = TILE;

  const int tid  = threadIdx.x;
  const int lane = tid & 31;
  const int wv   = tid >> 5;

  for (int r = tid; r < TILE; r += 256) {
    tl[r] = (r < n) ? bucket_t[row0 + r] : 0;
    wl[r] = (r < n) ? bucket_w[row0 + r] : 0.f;
  }
  __syncthreads();

  for (int i = tid; i < TILE * HD; i += 256) {   // gather token rows
    int r = i >> 7, d = i & 127;
    A[r * AST + d] = (r < n) ? (__bf16)x[(size_t)tl[r] * HD + d] : (__bf16)0.f;
  }
  const float* wge = wg + (size_t)e * HD * MI;
  const float* wue = wu + (size_t)e * HD * MI;
  const float* wde = wd + (size_t)e * MI * HD;
  for (int i = tid; i < HD * MI; i += 256) {     // [d][m] -> [m][d]
    int d = i / MI, mm = i - d * MI;
    Bg[mm * AST + d] = (__bf16)wge[i];
    Bu[mm * AST + d] = (__bf16)wue[i];
  }
  for (int i = tid; i < MI * HD; i += 256) {     // wd [m][d] -> [d][m]
    int mm = i >> 7, d = i & 127;
    Bd[d * MIP + mm] = (__bf16)wde[i];
  }
  for (int i = tid; i < HD * (MIP - MI); i += 256) {  // zero K-pad of Bd
    int d = i >> 4, mm = MI + (i & 15);
    Bd[d * MIP + mm] = (__bf16)0.f;
  }
  for (int i = tid; i < TILE * (MIP - MI); i += 256) { // zero K-pad of H
    int r = i >> 4, cpad = MI + (i & 15);
    H[r * MIP + cpad] = (__bf16)0.f;
  }
  __syncthreads();

  // GEMM1: H = silu(A@Wg)*(A@Wu) ; 4 m-tiles x 5 n-tiles, K=128 (4 steps)
  for (int task = wv; task < 20; task += 8) {
    int mt = task / 5, nt = task - mt * 5;
    v8f cg = {}; v8f cu = {};
    #pragma unroll
    for (int ks = 0; ks < 4; ks++) {
      v16bf a  = ld_frag_a(A,  mt * 16, ks * 32, AST, lane);
      v16bf bg = ld_frag_b(Bg, nt * 16, ks * 32, AST, lane);
      v16bf bu = ld_frag_b(Bu, nt * 16, ks * 32, AST, lane);
      cg = __builtin_amdgcn_wmma_f32_16x16x32_bf16(false, a, false, bg, (short)0, cg, false, false);
      cu = __builtin_amdgcn_wmma_f32_16x16x32_bf16(false, a, false, bu, (short)0, cu, false, false);
    }
    int rbase = mt * 16 + ((lane >> 4) << 3);
    int col   = nt * 16 + (lane & 15);
    #pragma unroll
    for (int r = 0; r < 8; r++)
      H[(rbase + r) * MIP + col] = (__bf16)(silu_f(cg[r]) * cu[r]);
  }
  __syncthreads();

  // GEMM2: Y = H @ Wd ; 4 m-tiles x 8 n-tiles, K=96 (3 steps); weighted atomic add
  for (int task = wv; task < 32; task += 8) {
    int mt = task >> 3, nt = task & 7;
    v8f c = {};
    #pragma unroll
    for (int ks = 0; ks < 3; ks++) {
      v16bf a = ld_frag_a(H,  mt * 16, ks * 32, MIP, lane);
      v16bf b = ld_frag_b(Bd, nt * 16, ks * 32, MIP, lane);
      c = __builtin_amdgcn_wmma_f32_16x16x32_bf16(false, a, false, b, (short)0, c, false, false);
    }
    int rbase = mt * 16 + ((lane >> 4) << 3);
    int col   = nt * 16 + (lane & 15);
    #pragma unroll
    for (int r = 0; r < 8; r++) {
      int mr = rbase + r;
      if (mr < n)
        atomicAdd(&out[(size_t)tl[mr] * HD + col], wl[mr] * c[r]);
    }
  }
}

// ---------------- host launcher ----------------
extern "C" void kernel_launch(void* const* d_in, const int* in_sizes, int n_in,
                              void* d_out, int out_size, void* d_ws, size_t ws_size,
                              hipStream_t stream) {
  const float* x      = (const float*)d_in[0];  // [B,S,D]
  const float* gate_w = (const float*)d_in[1];  // [E,D]
  const float* wg     = (const float*)d_in[2];  // [E,D,M]
  const float* wu     = (const float*)d_in[3];  // [E,D,M]
  const float* wd     = (const float*)d_in[4];  // [E,M,D]
  const float* swg    = (const float*)d_in[5];  // [D,MS]
  const float* swu    = (const float*)d_in[6];  // [D,MS]
  const float* swd    = (const float*)d_in[7];  // [MS,D]
  float* out = (float*)d_out;

  char* ws = (char*)d_ws;
  int*   counts    = (int*)(ws + 0);
  int*   offsets   = (int*)(ws + 256);
  int*   tileStart = (int*)(ws + 768);
  int*   cursor    = (int*)(ws + 1280);
  int*   topk_idx  = (int*)(ws + 2048);
  float* topk_w    = (float*)(ws + 2048 + NTOK * TK * 4);
  int*   bucket_t  = (int*)(ws + 2048 + NTOK * TK * 8);
  float* bucket_w  = (float*)(ws + 2048 + NTOK * TK * 12);

  k_init<<<1, 64, 0, stream>>>(counts);
  k_gate<<<NTOK / 4, 128, 0, stream>>>(x, gate_w, topk_idx, topk_w, counts);
  k_scan<<<1, 32, 0, stream>>>(counts, offsets, tileStart, cursor);
  k_scatter<<<(NTOK * TK + 255) / 256, 256, 0, stream>>>(topk_idx, topk_w, cursor,
                                                         bucket_t, bucket_w);
  // shared experts initialize d_out (plain stores, every element written)
  size_t lds_shared = (size_t)(TILE * HD + 2 * MS * HD + HD * MS + TILE * MS) * 2;
  k_shared<<<NTOK / TILE, 256, lds_shared, stream>>>(x, swg, swu, swd, out);
  // routed experts accumulate atomically on top
  size_t lds_routed = (size_t)(TILE * AST + 2 * MI * AST + HD * MIP + TILE * MIP) * 2
                    + TILE * 8;
  k_routed<<<MAXTILES, 256, lds_routed, stream>>>(x, wg, wu, wd, tileStart, offsets,
                                                  bucket_t, bucket_w, out);
}
